// _LongCatSingleAttention_76811195121708
// MI455X (gfx1250) — matrix operand order repaired and checked by reading, not compiled, not run on hardware
//
#include <hip/hip_runtime.h>

#define SEQ   2048
#define DIM   3072
#define NH    24
#define HD    128
#define INNER (NH*HD)

typedef __bf16 bf16_t;
typedef __attribute__((ext_vector_type(16))) __bf16 bf16x16;
typedef __attribute__((ext_vector_type(8)))  float  f32x8;

union BFrag { bf16x16 v; uint4 u[2]; };

// A-fragment (16x32 bf16): lane l -> row = l&15; for half = l>>4:
//   elements 0..15 = K[half*16 .. half*16+15]
__device__ __forceinline__ bf16x16 load_frag_a(const bf16_t* rowbase, int half) {
  BFrag t;
  t.u[0] = *(const uint4*)(rowbase + half * 16);
  t.u[1] = *(const uint4*)(rowbase + half * 16 + 8);
  return t.v;
}

// B-fragment (32x16 bf16): lane l -> col = l&15; elements 0..15 = K[(l>>4)*16 .. +15]
// caller passes pointer to the 16 consecutive K values for this lane.
__device__ __forceinline__ bf16x16 load_frag_b(const bf16_t* p) {
  BFrag t;
  const uint4* q = (const uint4*)p;
  t.u[0] = q[0];
  t.u[1] = q[1];
  return t.v;
}

__device__ __forceinline__ f32x8 wmma_bf16(bf16x16 a, bf16x16 b, f32x8 c) {
  return __builtin_amdgcn_wmma_f32_16x16x32_bf16(false, a, false, b, (short)0, c, false, false);
}

// ---------------------------------------------------------------- cvt f32->bf16
__global__ __launch_bounds__(256)
void cvt_f32_to_bf16(const float* __restrict__ src, bf16_t* __restrict__ dst, int n) {
  int i = (blockIdx.x * blockDim.x + threadIdx.x) * 8;
  if (i >= n) return;
  float4 a = *(const float4*)(src + i);
  float4 b = *(const float4*)(src + i + 4);
  union { bf16_t h[8]; uint4 u; } o;
  o.h[0] = (bf16_t)a.x; o.h[1] = (bf16_t)a.y; o.h[2] = (bf16_t)a.z; o.h[3] = (bf16_t)a.w;
  o.h[4] = (bf16_t)b.x; o.h[5] = (bf16_t)b.y; o.h[6] = (bf16_t)b.z; o.h[7] = (bf16_t)b.w;
  *(uint4*)(dst + i) = o.u;
}

// ---------------------------------------------------------------- QKV GEMM
// C[m,n] = sum_k hidden[m,k] * W[n,k] + bias[n]
// grid: (INNER/128, SEQ/128, 3); block 256 = 8 waves; wave tile 64(M) x 32(N).
// z=0 -> q (f32 [S,INNER]); z=1 -> k (f32); z=2 -> v stored bf16 transposed [INNER][SEQ].
__global__ __launch_bounds__(256)
void qkv_gemm(const bf16_t* __restrict__ hb,
              const bf16_t* __restrict__ wqb, const bf16_t* __restrict__ wkb,
              const bf16_t* __restrict__ wvb,
              const float* __restrict__ bq, const float* __restrict__ bk,
              const float* __restrict__ bv,
              float* __restrict__ qf, float* __restrict__ kf,
              bf16_t* __restrict__ vT)
{
  const int z = blockIdx.z;
  const bf16_t* wb   = (z == 0) ? wqb : (z == 1) ? wkb : wvb;
  const float*  bias = (z == 0) ? bq  : (z == 1) ? bk  : bv;

  const int lane = threadIdx.x & 31;
  const int wid  = threadIdx.x >> 5;
  const int wm   = wid >> 2, wn = wid & 3;
  const int row0 = blockIdx.y * 128 + wm * 64;
  const int col0 = blockIdx.x * 128 + wn * 32;
  const int r    = lane & 15, half = lane >> 4;

  f32x8 zero = {};
  f32x8 acc[4][2];
#pragma unroll
  for (int mi = 0; mi < 4; ++mi) { acc[mi][0] = zero; acc[mi][1] = zero; }

  const bf16_t* arow[4];
#pragma unroll
  for (int mi = 0; mi < 4; ++mi)
    arow[mi] = hb + (size_t)(row0 + mi * 16 + r) * DIM;
  const bf16_t* b0p = wb + (size_t)(col0 + r) * DIM + half * 16;
  const bf16_t* b1p = wb + (size_t)(col0 + 16 + r) * DIM + half * 16;

  for (int kk = 0; kk < DIM; kk += 32) {
    bf16x16 bf0 = load_frag_b(b0p + kk);
    bf16x16 bf1 = load_frag_b(b1p + kk);
#pragma unroll
    for (int mi = 0; mi < 4; ++mi) {
      bf16x16 af = load_frag_a(arow[mi] + kk, half);
      acc[mi][0] = wmma_bf16(af, bf0, acc[mi][0]);
      acc[mi][1] = wmma_bf16(af, bf1, acc[mi][1]);
    }
  }

  const float bias0 = bias[col0 + r];
  const float bias1 = bias[col0 + 16 + r];
  if (z < 2) {
    float* outp = (z == 0) ? qf : kf;
#pragma unroll
    for (int mi = 0; mi < 4; ++mi)
#pragma unroll
      for (int i = 0; i < 8; ++i) {
        const int row = row0 + mi * 16 + half * 8 + i;
        outp[(size_t)row * DIM + col0 + r]      = acc[mi][0][i] + bias0;
        outp[(size_t)row * DIM + col0 + 16 + r] = acc[mi][1][i] + bias1;
      }
  } else {
#pragma unroll
    for (int mi = 0; mi < 4; ++mi)
#pragma unroll
      for (int ni = 0; ni < 2; ++ni) {
        const float bb  = ni ? bias1 : bias0;
        const int   col = col0 + ni * 16 + r;
        union { bf16_t h[8]; uint4 u; } o;
#pragma unroll
        for (int i = 0; i < 8; ++i) o.h[i] = (bf16_t)(acc[mi][ni][i] + bb);
        *(uint4*)(vT + (size_t)col * SEQ + row0 + mi * 16 + half * 8) = o.u;
      }
  }
}

// ---------------------------------------------------------------- RMSNorm + RoPE + repack
// one wave per (s, h); lane holds d = lane*4 .. lane*4+3 (128 = 32*4).
__global__ __launch_bounds__(256)
void norm_rope_pack(const float* __restrict__ qf, const float* __restrict__ kf,
                    const float* __restrict__ nqw, const float* __restrict__ nkw,
                    const float* __restrict__ cosp, const float* __restrict__ sinp,
                    bf16_t* __restrict__ qbf, bf16_t* __restrict__ kbf)
{
  const int lane = threadIdx.x & 31;
  const int wid  = threadIdx.x >> 5;
  const int gw   = blockIdx.x * 8 + wid;      // [0, SEQ*NH)
  const int s    = gw / NH;
  const int h    = gw % NH;
  const int d0   = lane * 4;

  const float4 c4 = *(const float4*)(cosp + (size_t)s * HD + d0);
  const float4 s4 = *(const float4*)(sinp + (size_t)s * HD + d0);

#pragma unroll
  for (int which = 0; which < 2; ++which) {
    const float* src = which ? kf  : qf;
    const float* nw  = which ? nkw : nqw;
    bf16_t*      dst = which ? kbf : qbf;

    float4 x = *(const float4*)(src + (size_t)s * DIM + h * HD + d0);
    float ss = x.x * x.x + x.y * x.y + x.z * x.z + x.w * x.w;
#pragma unroll
    for (int m = 16; m >= 1; m >>= 1) ss += __shfl_xor(ss, m, 32);
    const float inv = rsqrtf(ss * (1.0f / HD) + 1e-6f);
    const float4 w4 = *(const float4*)(nw + d0);
    const float x0 = x.x * inv * w4.x, x1 = x.y * inv * w4.y;
    const float x2 = x.z * inv * w4.z, x3 = x.w * inv * w4.w;
    // interleaved RoPE pairs (d even, d odd): cos repeated per pair
    const float y0 = x0 * c4.x - x1 * s4.x;
    const float y1 = x1 * c4.y + x0 * s4.y;
    const float y2 = x2 * c4.z - x3 * s4.z;
    const float y3 = x3 * c4.w + x2 * s4.w;
    union { bf16_t h4[4]; uint2 u; } o;
    o.h4[0] = (bf16_t)y0; o.h4[1] = (bf16_t)y1;
    o.h4[2] = (bf16_t)y2; o.h4[3] = (bf16_t)y3;
    *(uint2*)(dst + ((size_t)h * SEQ + s) * HD + d0) = o.u;
  }
}

// ---------------------------------------------------------------- flash attention
// one wave per (head, 16-row q tile). K tiles of 32 keys. Full (non-causal) softmax.
__global__ __launch_bounds__(128)
void attn(const bf16_t* __restrict__ qbf, const bf16_t* __restrict__ kbf,
          const bf16_t* __restrict__ vT, float* __restrict__ out)
{
  __shared__ bf16_t pls[4][16][32];   // per-wave P staging for layout transpose

  const int lane = threadIdx.x & 31;
  const int wid  = threadIdx.x >> 5;
  const int gw   = blockIdx.x * 4 + wid;      // [0, NH*128)
  const int h    = gw >> 7;
  const int q0   = (gw & 127) * 16;
  const int r    = lane & 15, half = lane >> 4;

  const bf16_t* Q = qbf + (size_t)h * SEQ * HD;
  const bf16_t* K = kbf + (size_t)h * SEQ * HD;
  const bf16_t* V = vT  + (size_t)h * HD * SEQ;   // [d][s]

  bf16x16 qfrag[4];
#pragma unroll
  for (int dk = 0; dk < 4; ++dk)
    qfrag[dk] = load_frag_a(Q + (size_t)(q0 + r) * HD + dk * 32, half);

  f32x8 zero = {};
  f32x8 O[8];
#pragma unroll
  for (int t = 0; t < 8; ++t) O[t] = zero;
  float mrow[8], lrow[8];
#pragma unroll
  for (int i = 0; i < 8; ++i) { mrow[i] = -1e30f; lrow[i] = 0.f; }
  const float scale = 0.08838834764831845f;   // 1/sqrt(128)

  for (int kt = 0; kt < SEQ; kt += 32) {
    // S tiles: rows q0..q0+15, key cols kt..kt+15 (s0) and kt+16..kt+31 (s1)
    f32x8 s0 = zero, s1 = zero;
#pragma unroll
    for (int dk = 0; dk < 4; ++dk) {
      bf16x16 k0 = load_frag_b(K + (size_t)(kt + r) * HD + dk * 32 + half * 16);
      bf16x16 k1 = load_frag_b(K + (size_t)(kt + 16 + r) * HD + dk * 32 + half * 16);
      s0 = wmma_bf16(qfrag[dk], k0, s0);
      s1 = wmma_bf16(qfrag[dk], k1, s1);
    }
    // online softmax update; lane holds column r for rows half*8+i
#pragma unroll
    for (int i = 0; i < 8; ++i) {
      const float a = s0[i] * scale;
      const float b = s1[i] * scale;
      float mx = fmaxf(a, b);
      mx = fmaxf(mx, __shfl_xor(mx, 1, 32));
      mx = fmaxf(mx, __shfl_xor(mx, 2, 32));
      mx = fmaxf(mx, __shfl_xor(mx, 4, 32));
      mx = fmaxf(mx, __shfl_xor(mx, 8, 32));
      const float mnew  = fmaxf(mrow[i], mx);
      const float alpha = __expf(mrow[i] - mnew);
      const float p0 = __expf(a - mnew);
      const float p1 = __expf(b - mnew);
      float rs = p0 + p1;
      rs += __shfl_xor(rs, 1, 32);
      rs += __shfl_xor(rs, 2, 32);
      rs += __shfl_xor(rs, 4, 32);
      rs += __shfl_xor(rs, 8, 32);
      lrow[i] = lrow[i] * alpha + rs;
      mrow[i] = mnew;
#pragma unroll
      for (int t = 0; t < 8; ++t) O[t][i] *= alpha;
      pls[wid][half * 8 + i][r]      = (bf16_t)p0;
      pls[wid][half * 8 + i][r + 16] = (bf16_t)p1;
    }
    // reload P in A-fragment layout (per-wave LDS region; in-order DS, no barrier)
    bf16x16 pf = load_frag_a(&pls[wid][r][0], half);
#pragma unroll
    for (int dt = 0; dt < 8; ++dt) {
      bf16x16 vf = load_frag_b(V + (size_t)(dt * 16 + r) * SEQ + kt + half * 16);
      O[dt] = wmma_bf16(pf, vf, O[dt]);
    }
  }

  float invl[8];
#pragma unroll
  for (int i = 0; i < 8; ++i) invl[i] = 1.0f / lrow[i];
#pragma unroll
  for (int dt = 0; dt < 8; ++dt)
#pragma unroll
    for (int i = 0; i < 8; ++i)
      out[(size_t)(q0 + half * 8 + i) * DIM + h * HD + dt * 16 + r] = O[dt][i] * invl[i];
}

// ---------------------------------------------------------------- launcher
extern "C" void kernel_launch(void* const* d_in, const int* in_sizes, int n_in,
                              void* d_out, int out_size, void* d_ws, size_t ws_size,
                              hipStream_t stream) {
  const float* hidden = (const float*)d_in[0];
  const float* wq  = (const float*)d_in[1];
  const float* bq  = (const float*)d_in[2];
  const float* wk  = (const float*)d_in[3];
  const float* bk  = (const float*)d_in[4];
  const float* wv  = (const float*)d_in[5];
  const float* bv  = (const float*)d_in[6];
  const float* nqw = (const float*)d_in[7];
  const float* nkw = (const float*)d_in[8];
  const float* cosp = (const float*)d_in[9];
  const float* sinp = (const float*)d_in[10];
  float* out = (float*)d_out;

  char* ws = (char*)d_ws;
  bf16_t* hb  = (bf16_t*)(ws);                    // 12,582,912  hidden bf16
  bf16_t* wqb = (bf16_t*)(ws + 12582912);         // 18,874,368  wq bf16
  bf16_t* wkb = (bf16_t*)(ws + 31457280);         // 18,874,368
  bf16_t* wvb = (bf16_t*)(ws + 50331648);         // 18,874,368
  float*  qf  = (float* )(ws + 69206016);         // 25,165,824  q f32
  float*  kf  = (float* )(ws + 94371840);         // 25,165,824
  bf16_t* qbf = (bf16_t*)(ws + 119537664);        // 12,582,912  q bf16 [H][S][D]
  bf16_t* kbf = (bf16_t*)(ws + 132120576);        // 12,582,912
  bf16_t* vT  = (bf16_t*)(ws + 144703488);        // 12,582,912  v bf16 [H*D][S]

  cvt_f32_to_bf16<<<(SEQ * DIM) / 2048, 256, 0, stream>>>(hidden, hb, SEQ * DIM);
  cvt_f32_to_bf16<<<(DIM * DIM) / 2048, 256, 0, stream>>>(wq, wqb, DIM * DIM);
  cvt_f32_to_bf16<<<(DIM * DIM) / 2048, 256, 0, stream>>>(wk, wkb, DIM * DIM);
  cvt_f32_to_bf16<<<(DIM * DIM) / 2048, 256, 0, stream>>>(wv, wvb, DIM * DIM);

  qkv_gemm<<<dim3(INNER / 128, SEQ / 128, 3), 256, 0, stream>>>(
      hb, wqb, wkb, wvb, bq, bk, bv, qf, kf, vT);

  norm_rope_pack<<<(SEQ * NH) / 8, 256, 0, stream>>>(
      qf, kf, nqw, nkw, cosp, sinp, qbf, kbf);

  attn<<<(NH * (SEQ / 16)) / 4, 128, 0, stream>>>(qbf, kbf, vT, out);
}
